// Memory_12592844112347
// MI455X (gfx1250) — matrix operand order repaired and checked by reading, not compiled, not run on hardware
//
#include <hip/hip_runtime.h>
#include <hip/hip_bf16.h>

typedef __attribute__((ext_vector_type(16))) __bf16 v16bf;
typedef __attribute__((ext_vector_type(8)))  float  v8f;
typedef unsigned short u16;

#define TEMP_INV 10.0f
#define EPS_LOG  1e-6f
#define RMS_EPS  1e-6f
#define KNEI 8
#define BT   4096      // B*T
#define DDIM 1024      // D
#define NKEY 32768     // N memory slots
#define DK   256
#define DM   256
#define FDIM 2048      // KNEI*DM

// ---------- helpers ----------

__device__ __forceinline__ u16 f2bf(float f) {   // f32 -> bf16, round-nearest-even
  unsigned u = __float_as_uint(f);
  u += 0x7FFFu + ((u >> 16) & 1u);
  return (u16)(u >> 16);
}

// WMMA A-matrix (16x32 bf16) register layout used by this code (contiguous k, same for B):
// lanes 0-15: M=lane, V0..V7 -> K=0..15
// lanes16-31: M=lane-16,      K=16..31
// Returns u32 index into a 512-entry u16 LDS tile for packed pair (K even).
__device__ __forceinline__ int a_idx_u32(int rm, int kk) {
  int half = kk >> 4;
  int idx  = kk & 15;
  return (half * 16 + rm) * 8 + (idx >> 1);
}

// Stage a 16x32 tile (bf16 source) into LDS in A layout. 256 threads, 2 elems each.
__device__ __forceinline__ void stageA_bf16(u16* ldsA, const u16* src, int stride, int tid) {
  int rm = tid >> 4, kk = (tid & 15) << 1;
  unsigned v = *reinterpret_cast<const unsigned*>(src + rm * stride + kk);
  reinterpret_cast<unsigned*>(ldsA)[a_idx_u32(rm, kk)] = v;
}

// Same, converting from f32 source on the fly.
__device__ __forceinline__ void stageA_f32(u16* ldsA, const float* src, int stride, int tid) {
  int rm = tid >> 4, kk = (tid & 15) << 1;
  float2 f = *reinterpret_cast<const float2*>(src + rm * stride + kk);
  reinterpret_cast<unsigned*>(ldsA)[a_idx_u32(rm, kk)] =
      (unsigned)f2bf(f.x) | ((unsigned)f2bf(f.y) << 16);
}

__device__ __forceinline__ v8f wmma_bf16(v16bf a, v16bf b, v8f c) {
  return __builtin_amdgcn_wmma_f32_16x16x32_bf16(false, a, false, b, (short)0, c,
                                                 false, false);
}

// ---------- prep kernels ----------

// keys f32 -> bf16 (plain [N][DK] layout, lane-contiguous for WMMA-B) + k2[n]=||k||^2
__global__ __launch_bounds__(256) void k_prep_keys(const float* __restrict__ keys,
                                                   u16* __restrict__ keysb,
                                                   float* __restrict__ k2) {
  int row = blockIdx.x * 8 + (threadIdx.x >> 5);
  int ln  = threadIdx.x & 31;
  const float* kr = keys + (size_t)row * DK;
  float s = 0.f;
#pragma unroll
  for (int i = 0; i < 8; ++i) {
    float v = kr[ln + i * 32];
    keysb[(size_t)row * DK + ln + i * 32] = f2bf(v);
    s += v * v;
  }
#pragma unroll
  for (int m = 16; m >= 1; m >>= 1) s += __shfl_xor(s, m, 32);
  if (ln == 0) k2[row] = s;
}

__global__ __launch_bounds__(256) void k_cvt_bf16(const float* __restrict__ src,
                                                  u16* __restrict__ dst, int n) {
  int i = blockIdx.x * 256 + threadIdx.x;
  if (i < n) dst[i] = f2bf(src[i]);
}

// vals f32 [N][DM] -> bf16 pre-shuffled into exact WMMA B-register order:
// flat e = (((kt*16 + g)*32) + ln)*16 + j  holds vals[kt*32 + (ln>>4)*16 + j][g*16 + (ln&15)]
__global__ __launch_bounds__(256) void k_shuffle_vals(const float* __restrict__ vals,
                                                      u16* __restrict__ valsB) {
  int e = blockIdx.x * 256 + threadIdx.x;   // 0 .. N*DM-1 (exact multiple)
  int j = e & 15, ln = (e >> 4) & 31, g = (e >> 9) & 15, kt = e >> 13;
  int srow = kt * 32 + (ln >> 4) * 16 + j;
  int scol = g * 16 + (ln & 15);
  valsB[e] = f2bf(vals[(size_t)srow * DM + scol]);
}

// ---------- q = x @ W_in^T + b_in ; also q2 = ||q||^2 ----------

__global__ __launch_bounds__(256) void k_proj(const float* __restrict__ x,
                                              const u16* __restrict__ winb,
                                              const float* __restrict__ b_in,
                                              u16* __restrict__ qbf,
                                              float* __restrict__ q2) {
  __shared__ __align__(32) u16 ldsA[512];
  __shared__ float q2s[16];
  int tid = threadIdx.x, wave = tid >> 5, ln = tid & 31;
  int rowbase = blockIdx.x * 16;
  if (tid < 16) q2s[tid] = 0.f;
  v8f acc0 = {0.f,0.f,0.f,0.f,0.f,0.f,0.f,0.f};
  v8f acc1 = acc0;
  for (int kt = 0; kt < DDIM / 32; ++kt) {
    __syncthreads();
    stageA_f32(ldsA, x + (size_t)rowbase * DDIM + kt * 32, DDIM, tid);
    __syncthreads();
    v16bf a = *reinterpret_cast<const v16bf*>(ldsA + ln * 16);
    const u16* bp = winb + (size_t)(wave * 32 + (ln & 15)) * DDIM + kt * 32 + (ln >> 4) * 16;
    acc0 = wmma_bf16(a, *reinterpret_cast<const v16bf*>(bp), acc0);
    acc1 = wmma_bf16(a, *reinterpret_cast<const v16bf*>(bp + 16 * DDIM), acc1);
  }
  int half = ln >> 4, cl = ln & 15;
#pragma unroll
  for (int v = 0; v < 8; ++v) {
    int row = v + 8 * half;
    int c0 = wave * 32 + cl, c1 = c0 + 16;
    float v0 = acc0[v] + b_in[c0];
    float v1 = acc1[v] + b_in[c1];
    qbf[(size_t)(rowbase + row) * DK + c0] = f2bf(v0);
    qbf[(size_t)(rowbase + row) * DK + c1] = f2bf(v1);
    atomicAdd(&q2s[row], v0 * v0 + v1 * v1);
  }
  __syncthreads();
  if (tid < 16) q2[rowbase + tid] = q2s[tid];
}

// ---------- logits = (2*q.k - q2 - k2) / TEMP ----------

__global__ __launch_bounds__(256) void k_logits(const u16* __restrict__ qbf,
                                                const u16* __restrict__ keysb,
                                                const float* __restrict__ q2,
                                                const float* __restrict__ k2,
                                                float* __restrict__ lg) {
  __shared__ __align__(32) u16 ldsA[512];
  __shared__ float q2s[16];
  __shared__ float k2s[256];
  int tid = threadIdx.x, wave = tid >> 5, ln = tid & 31;
  int rowbase = blockIdx.x * 16;
  int colbase = blockIdx.y * 256;
  if (tid < 16) q2s[tid] = q2[rowbase + tid];
  k2s[tid] = k2[colbase + tid];
  v8f acc0 = {0.f,0.f,0.f,0.f,0.f,0.f,0.f,0.f};
  v8f acc1 = acc0;
  for (int kt = 0; kt < DK / 32; ++kt) {
    __syncthreads();
    stageA_bf16(ldsA, qbf + (size_t)rowbase * DK + kt * 32, DK, tid);
    __syncthreads();
    v16bf a = *reinterpret_cast<const v16bf*>(ldsA + ln * 16);
    const u16* bp = keysb + (size_t)(colbase + wave * 32 + (ln & 15)) * DK + kt * 32 + (ln >> 4) * 16;
    acc0 = wmma_bf16(a, *reinterpret_cast<const v16bf*>(bp), acc0);
    acc1 = wmma_bf16(a, *reinterpret_cast<const v16bf*>(bp + 16 * DK), acc1);
  }
  int half = ln >> 4, cl = ln & 15;
#pragma unroll
  for (int v = 0; v < 8; ++v) {
    int row = v + 8 * half;
    int c0 = wave * 32 + cl, c1 = c0 + 16;
    size_t base = (size_t)(rowbase + row) * NKEY + colbase;
    lg[base + c0] = (2.f * acc0[v] - q2s[row] - k2s[c0]) * TEMP_INV;
    lg[base + c1] = (2.f * acc1[v] - q2s[row] - k2s[c1]) * TEMP_INV;
  }
}

// ---------- one k-NN round: lazy logit update + online softmax + p@vals (WMMA) ----------

__global__ __launch_bounds__(256) void k_round(float* __restrict__ lg,
                                               const u16* __restrict__ valsB,
                                               u16* __restrict__ nearestb,
                                               float* __restrict__ sm,
                                               float* __restrict__ sS, int r) {
  __shared__ __align__(32) u16 ldsA[512];
  __shared__ float mrun[16], Srun[16], mprev[16], iSprev[16], tmax[16], tsum[16], fact[16];
  int tid = threadIdx.x, wave = tid >> 5, ln = tid & 31;
  int rowbase = blockIdx.x * 16;
  if (tid < 16) {
    mrun[tid] = -3.0e38f;
    Srun[tid] = 0.f;
    if (r > 0) { mprev[tid] = sm[rowbase + tid]; iSprev[tid] = 1.f / sS[rowbase + tid]; }
  }
  v8f acc0 = {0.f,0.f,0.f,0.f,0.f,0.f,0.f,0.f};
  v8f acc1 = acc0;
  int rm = tid >> 4, kk = (tid & 15) << 1;
  float* lgrow = lg + (size_t)(rowbase + rm) * NKEY + kk;
  __syncthreads();
  for (int kt = 0; kt < NKEY / 32; ++kt) {
    float2 lv = *reinterpret_cast<float2*>(lgrow + kt * 32);
    if (r > 0) {  // apply previous round's  lg += log1p(EPS - w_prev)
      float w0 = __expf(lv.x - mprev[rm]) * iSprev[rm];
      float w1 = __expf(lv.y - mprev[rm]) * iSprev[rm];
      lv.x += __logf(1.0f - w0 + EPS_LOG);
      lv.y += __logf(1.0f - w1 + EPS_LOG);
      if (r < KNEI - 1) *reinterpret_cast<float2*>(lgrow + kt * 32) = lv;
    }
    // tile row-max (16 lanes per row)
    float tm = fmaxf(lv.x, lv.y);
#pragma unroll
    for (int m = 8; m >= 1; m >>= 1) tm = fmaxf(tm, __shfl_xor(tm, m, 32));
    if ((tid & 15) == 0) tmax[rm] = tm;
    __syncthreads();
    if (tid < 16) {
      float mn = fmaxf(mrun[tid], tmax[tid]);
      fact[tid] = __expf(mrun[tid] - mn);
      mrun[tid] = mn;
    }
    __syncthreads();
    // p = exp(lg - m_run), packed to bf16 into A-layout LDS
    float p0 = __expf(lv.x - mrun[rm]);
    float p1 = __expf(lv.y - mrun[rm]);
    reinterpret_cast<unsigned*>(ldsA)[a_idx_u32(rm, kk)] =
        (unsigned)f2bf(p0) | ((unsigned)f2bf(p1) << 16);
    float ps = p0 + p1;
#pragma unroll
    for (int m = 8; m >= 1; m >>= 1) ps += __shfl_xor(ps, m, 32);
    if ((tid & 15) == 0) tsum[rm] = ps;
    __syncthreads();
    if (tid < 16) Srun[tid] = Srun[tid] * fact[tid] + tsum[tid];
    // rescale accumulators (flash-style) and do the tile GEMM
    int halfL = ln >> 4;
#pragma unroll
    for (int v = 0; v < 8; ++v) {
      float f = fact[v + 8 * halfL];
      acc0[v] *= f;
      acc1[v] *= f;
    }
    v16bf a = *reinterpret_cast<const v16bf*>(ldsA + ln * 16);
    const u16* bp = valsB + (size_t)(kt * 16 + wave * 2) * 512 + ln * 16;
    acc0 = wmma_bf16(a, *reinterpret_cast<const v16bf*>(bp), acc0);
    acc1 = wmma_bf16(a, *reinterpret_cast<const v16bf*>(bp + 512), acc1);
    if (kt + 1 < NKEY / 32)
      __builtin_prefetch(valsB + (size_t)((kt + 1) * 16 + wave * 2) * 512 + ln * 16, 0, 1);
    __syncthreads();
  }
  // epilogue: normalize by S and store bf16 slice of 'nearest'
  int halfL = ln >> 4, cl = ln & 15;
#pragma unroll
  for (int v = 0; v < 8; ++v) {
    int row = v + 8 * halfL;
    float inv = 1.0f / Srun[row];
    int c0 = wave * 32 + cl;
    size_t base = (size_t)(rowbase + row) * FDIM + r * DM;
    nearestb[base + c0]      = f2bf(acc0[v] * inv);
    nearestb[base + c0 + 16] = f2bf(acc1[v] * inv);
  }
  if (tid < 16 && r < KNEI - 1) {
    sm[rowbase + tid] = mrun[tid];
    sS[rowbase + tid] = Srun[tid];
  }
}

// ---------- out = RMSNorm(nearest @ W_out^T + b_out) * rms_w ----------

__global__ __launch_bounds__(256) void k_out(const u16* __restrict__ nearestb,
                                             const u16* __restrict__ woutb,
                                             const float* __restrict__ b_out,
                                             const float* __restrict__ rms_w,
                                             float* __restrict__ out) {
  __shared__ __align__(32) u16 ldsA[512];
  __shared__ float ss[16];
  int tid = threadIdx.x, wave = tid >> 5, ln = tid & 31;
  int rowbase = blockIdx.x * 16;
  if (tid < 16) ss[tid] = 0.f;
  v8f acc[8] = {};   // 8 tiles of 16 cols -> wave covers 128 of 1024 cols
  for (int kt = 0; kt < FDIM / 32; ++kt) {
    __syncthreads();
    stageA_bf16(ldsA, nearestb + (size_t)rowbase * FDIM + kt * 32, FDIM, tid);
    __syncthreads();
    v16bf a = *reinterpret_cast<const v16bf*>(ldsA + ln * 16);
#pragma unroll
    for (int g = 0; g < 8; ++g) {
      const u16* bp = woutb + (size_t)(wave * 128 + g * 16 + (ln & 15)) * FDIM +
                      kt * 32 + (ln >> 4) * 16;
      acc[g] = wmma_bf16(a, *reinterpret_cast<const v16bf*>(bp), acc[g]);
    }
  }
  __syncthreads();
  int halfL = ln >> 4, cl = ln & 15;
#pragma unroll
  for (int v = 0; v < 8; ++v) {
    float part = 0.f;
#pragma unroll
    for (int g = 0; g < 8; ++g) {
      int col = wave * 128 + g * 16 + cl;
      float val = acc[g][v] + b_out[col];
      acc[g][v] = val;
      part += val * val;
    }
#pragma unroll
    for (int m = 8; m >= 1; m >>= 1) part += __shfl_xor(part, m, 32);
    if (cl == 0) atomicAdd(&ss[v + 8 * halfL], part);
  }
  __syncthreads();
  if (tid < 16) ss[tid] = rsqrtf(ss[tid] * (1.0f / (float)DDIM) + RMS_EPS);
  __syncthreads();
#pragma unroll
  for (int v = 0; v < 8; ++v) {
    int row = v + 8 * halfL;
    float sc = ss[row];
#pragma unroll
    for (int g = 0; g < 8; ++g) {
      int col = wave * 128 + g * 16 + cl;
      out[(size_t)(rowbase + row) * DDIM + col] = acc[g][v] * sc * rms_w[col];
    }
  }
}

// ---------- host ----------

extern "C" void kernel_launch(void* const* d_in, const int* in_sizes, int n_in,
                              void* d_out, int out_size, void* d_ws, size_t ws_size,
                              hipStream_t stream) {
  const float* x     = (const float*)d_in[0];
  const float* keys  = (const float*)d_in[1];
  const float* vals  = (const float*)d_in[2];
  const float* W_in  = (const float*)d_in[3];
  const float* b_in  = (const float*)d_in[4];
  const float* W_out = (const float*)d_in[5];
  const float* b_out = (const float*)d_in[6];
  const float* rms_w = (const float*)d_in[7];
  float* out = (float*)d_out;

  char* w = (char*)d_ws;
  size_t off = 0;
  auto alloc = [&](size_t bytes) {
    void* p = w + off;
    off = (off + bytes + 255) & ~(size_t)255;
    return p;
  };
  float* lg       = (float*)alloc((size_t)BT * NKEY * 4);   // 512 MB
  u16*   valsB    = (u16*)  alloc((size_t)NKEY * DM * 2);   // 16 MB (B-layout)
  u16*   keysb    = (u16*)  alloc((size_t)NKEY * DK * 2);   // 16 MB
  u16*   winb     = (u16*)  alloc((size_t)DK * DDIM * 2);
  u16*   woutb    = (u16*)  alloc((size_t)DDIM * FDIM * 2);
  u16*   qbf      = (u16*)  alloc((size_t)BT * DK * 2);
  u16*   nearestb = (u16*)  alloc((size_t)BT * FDIM * 2);   // 16 MB
  float* q2       = (float*)alloc((size_t)BT * 4);
  float* k2       = (float*)alloc((size_t)NKEY * 4);
  float* sm       = (float*)alloc((size_t)BT * 4);
  float* sS       = (float*)alloc((size_t)BT * 4);

  k_prep_keys<<<NKEY / 8, 256, 0, stream>>>(keys, keysb, k2);
  k_cvt_bf16<<<(DK * DDIM + 255) / 256, 256, 0, stream>>>(W_in, winb, DK * DDIM);
  k_cvt_bf16<<<(DDIM * FDIM + 255) / 256, 256, 0, stream>>>(W_out, woutb, DDIM * FDIM);
  k_shuffle_vals<<<(NKEY * DM) / 256, 256, 0, stream>>>(vals, valsB);
  k_proj<<<BT / 16, 256, 0, stream>>>(x, winb, b_in, qbf, q2);
  dim3 lgrid(BT / 16, NKEY / 256);
  k_logits<<<lgrid, 256, 0, stream>>>(qbf, keysb, q2, k2, lg);
  for (int r = 0; r < KNEI; ++r)
    k_round<<<BT / 16, 256, 0, stream>>>(lg, valsB, nearestb, sm, sS, r);
  k_out<<<BT / 16, 256, 0, stream>>>(nearestb, woutb, b_out, rms_w, out);
}
